// PolarTransformAttention_90494960927274
// MI455X (gfx1250) — matrix operand order repaired and checked by reading, not compiled, not run on hardware
//
#include <hip/hip_runtime.h>
#include <hip/hip_bf16.h>
#include <math.h>

// ---------------------------------------------------------------------------
// PolarTransformAttention, fused for MI455X (gfx1250).
//
// Shapes: x (16,256,128,128) f32; pf/pp (16,256,16,8); q (16,64,128);
//         att (16,128,128); out = x * att.
//
// Key fusion: polar_to_cartesian is linear in pp, so the 268 MB pc tensor is
// never materialized: q = w_attn1 @ pp (tiny WMMA GEMM), then per-pixel
// bilinear(q) + relu + w_attn2 + sigmoid gives att (1 MB). Only unavoidable
// HBM traffic is read x + write out (~536 MB -> ~23 us at 23.3 TB/s).
//
//   K1  cart2polar   : x -> pf                     (gather, tiny)
//   K2  polar_conv   : pf -> pp  (grouped 3x3)     (LDS-staged VALU)
//   K3  attn1 GEMM   : q[b] = w_attn1 @ pp[b]      (WMMA bf16, 8x unrolled)
//   K4  att map      : per-pixel bilinear(q)+relu+dot(w2)+sigmoid -> att
//   K5  modulate     : out = x * att               (streaming, NT hints)
// ---------------------------------------------------------------------------

typedef __attribute__((ext_vector_type(16))) __bf16 v16bf;
typedef __attribute__((ext_vector_type(8)))  float  v8f;
typedef __attribute__((ext_vector_type(4)))  float  f4;

#define B_  16
#define C_  256
#define H_  128
#define W_  128
#define AB_ 16      // angle bins
#define RB_ 8       // radius bins
#define P_  (AB_*RB_)      // 128 polar positions
#define O_  64             // attn hidden channels
#define TWO_PI 6.28318530717958647692f

// --------------------------- K1: cartesian -> polar -------------------------
__global__ __launch_bounds__(256) void k1_cart2polar(
    const float* __restrict__ x, float* __restrict__ pf)
{
  int tid = blockIdx.x * 256 + threadIdx.x;            // B*C*P = 524288
  if (tid >= B_ * C_ * P_) return;
  int p  = tid & (P_ - 1);
  int bc = tid >> 7;                                   // b*256 + c
  int a  = p >> 3, r = p & 7;

  float theta = TWO_PI * (float)a / (float)AB_;
  float rad   = ((float)r + 0.5f) / (float)RB_ * 64.0f;   // max_r = 64
  float ys = 63.5f + rad * __sinf(theta);
  float xs = 63.5f + rad * __cosf(theta);

  float y0f = floorf(ys), x0f = floorf(xs);
  float wy = ys - y0f, wx = xs - x0f;
  int y0i = min(max((int)y0f,     0), H_ - 1);
  int y1i = min(max((int)y0f + 1, 0), H_ - 1);
  int x0i = min(max((int)x0f,     0), W_ - 1);
  int x1i = min(max((int)x0f + 1, 0), W_ - 1);

  const float* plane = x + (size_t)bc * (H_ * W_);
  float v00 = plane[y0i * W_ + x0i];
  float v01 = plane[y0i * W_ + x1i];
  float v10 = plane[y1i * W_ + x0i];
  float v11 = plane[y1i * W_ + x1i];
  float out = v00 * (1.f - wy) * (1.f - wx) + v01 * (1.f - wy) * wx
            + v10 * wy * (1.f - wx)         + v11 * wy * wx;
  float valid = (ys >= 0.f && ys <= (float)(H_ - 1) &&
                 xs >= 0.f && xs <= (float)(W_ - 1)) ? 1.f : 0.f;
  pf[tid] = out * valid;
}

// ----------------- K2: grouped 3x3 conv on polar grid (16x8) ----------------
// feature_group_count = 4; group g: out ch [g*64, g*64+64), in ch same range.
__global__ __launch_bounds__(256) void k2_polar_conv(
    const float* __restrict__ pf, const float* __restrict__ wp,
    const float* __restrict__ bp, float* __restrict__ pp)
{
  int g = blockIdx.x & 3;
  int b = blockIdx.x >> 2;
  __shared__ float s[64 * P_];                         // 32 KB input slab
  const float* src = pf + (size_t)b * (C_ * P_) + (size_t)g * 64 * P_;
  for (int i = threadIdx.x; i < 64 * P_; i += 256) s[i] = src[i];
  __syncthreads();

  int o_local = threadIdx.x & 63;
  int pq      = threadIdx.x >> 6;                      // quarter of the 128 positions
  int o       = g * 64 + o_local;
  const float* wrow = wp + (size_t)o * (64 * 9);

  float acc[32];
  float bias = bp[o];
#pragma unroll
  for (int j = 0; j < 32; ++j) acc[j] = bias;

  for (int i = 0; i < 64; ++i) {
    float w9[9];
#pragma unroll
    for (int t = 0; t < 9; ++t) w9[t] = wrow[i * 9 + t];
    const float* si = s + i * P_;
#pragma unroll
    for (int j = 0; j < 32; ++j) {
      int p = pq * 32 + j;
      int a = p >> 3, r = p & 7;
      float v = 0.f;
#pragma unroll
      for (int dy = 0; dy < 3; ++dy) {
        int aa = a + dy - 1;
        if (aa < 0 || aa > AB_ - 1) continue;
#pragma unroll
        for (int dx = 0; dx < 3; ++dx) {
          int rr = r + dx - 1;
          if (rr < 0 || rr > RB_ - 1) continue;
          v = fmaf(w9[dy * 3 + dx], si[aa * 8 + rr], v);
        }
      }
      acc[j] += v;
    }
  }
  float* dst = pp + (size_t)b * (C_ * P_) + (size_t)o * P_ + pq * 32;
#pragma unroll
  for (int j = 0; j < 32; ++j) dst[j] = acc[j];
}

// -------- K3: q[b] (64x128) = w_attn1 (64x256) @ pp[b] (256x128), WMMA ------
// One wave per 16x16 C/D tile. K=256 -> 8 x v_wmma_f32_16x16x32_bf16, fully
// unrolled so fragment loads + bf16 packs pipeline across WMMA issues (the
// only cross-WMMA dependency is the accumulator C->D chain).
// A layout (16-bit 16x32): lane half h, row m=lane&15:
//   elems 0..15 -> K = k0 + h*16 + e
// B layout (32x16): lanes 0-15: N=lane, K=k0+e ; lanes 16-31: N=lane-16, K=k0+16+e
// C/D: lane half h, col n=lane&15; VGPR v -> row M = h*8 + v.
__global__ __launch_bounds__(32) void k3_attn1_wmma(
    const float* __restrict__ pp, const float* __restrict__ w1,
    float* __restrict__ q)
{
  int tile = blockIdx.x;                    // B * 4 * 8 tiles
  int nt = tile & 7;  tile >>= 3;
  int mt = tile & 3;  int b = tile >> 2;

  int lane = threadIdx.x;
  int half = lane >> 4;
  int l    = lane & 15;

  const float* arow = w1 + (size_t)(mt * 16 + l) * C_;      // A row M
  const float* ppb  = pp + (size_t)b * (C_ * P_);
  int n = nt * 16 + l;                                      // B/D column

  v8f acc = {};
#pragma unroll
  for (int k0 = 0; k0 < C_; k0 += 32) {
    v16bf av, bv;
#pragma unroll
    for (int e = 0; e < 8; ++e) {
      av[e]     = (__bf16)arow[k0 + half * 16 + e];
      av[e + 8] = (__bf16)arow[k0 + half * 16 + 8 + e];
    }
#pragma unroll
    for (int e = 0; e < 16; ++e) {
      int k = k0 + half * 16 + e;
      bv[e] = (__bf16)ppb[(size_t)k * P_ + n];
    }
    acc = __builtin_amdgcn_wmma_f32_16x16x32_bf16(
        /*neg_a=*/false, av, /*neg_b=*/false, bv,
        /*c_mod=*/(short)0, acc, /*reuse_a=*/false, /*reuse_b=*/false);
  }

  float* qb = q + (size_t)b * (O_ * P_) + (size_t)(mt * 16) * P_ + n;
#pragma unroll
  for (int v = 0; v < 8; ++v) {
    int row = half * 8 + v;
    qb[row * P_] = acc[v];
  }
}

// ------------- K4: per-pixel polar resample + relu + attn2 + sigmoid --------
__global__ __launch_bounds__(256) void k4_att(
    const float* __restrict__ q, const float* __restrict__ b1,
    const float* __restrict__ w2, const float* __restrict__ b2,
    float* __restrict__ att)
{
  int b   = blockIdx.x >> 6;
  int blk = blockIdx.x & 63;
  __shared__ float sq[O_ * P_];             // 32 KB
  __shared__ float sb1[O_], sw2[O_];
  const float* qb = q + (size_t)b * (O_ * P_);
  for (int i = threadIdx.x; i < O_ * P_; i += 256) sq[i] = qb[i];
  if (threadIdx.x < O_) { sb1[threadIdx.x] = b1[threadIdx.x];
                          sw2[threadIdx.x] = w2[threadIdx.x]; }
  __syncthreads();

  int pix = blk * 256 + threadIdx.x;        // 0..16383
  int h = pix >> 7, w = pix & 127;
  float dy = (float)h - 63.5f, dx = (float)w - 63.5f;
  float rr = sqrtf(dy * dy + dx * dx);
  float th = atan2f(dy, dx);
  if (th < 0.f) th += TWO_PI;
  float a  = th * ((float)AB_ / TWO_PI);
  float ri = rr * ((float)RB_ / 64.0f) - 0.5f;
  float a0 = floorf(a);
  float wa = a - a0;
  int a0i = ((int)a0) & (AB_ - 1);
  int a1i = (a0i + 1) & (AB_ - 1);
  float r0 = floorf(ri);
  float wr = ri - r0;
  int r0i = min(max((int)r0,     0), RB_ - 1);
  int r1i = min(max((int)r0 + 1, 0), RB_ - 1);
  float valid = (rr <= 64.0f) ? 1.f : 0.f;

  float w00 = (1.f - wa) * (1.f - wr), w01 = (1.f - wa) * wr;
  float w10 = wa * (1.f - wr),          w11 = wa * wr;
  int i00 = a0i * 8 + r0i, i01 = a0i * 8 + r1i;
  int i10 = a1i * 8 + r0i, i11 = a1i * 8 + r1i;

  float acc = 0.f;
#pragma unroll 8
  for (int o = 0; o < O_; ++o) {
    const float* qo = sq + o * P_;
    float s = qo[i00] * w00 + qo[i01] * w01 + qo[i10] * w10 + qo[i11] * w11;
    float h1 = fmaxf(s * valid + sb1[o], 0.f);
    acc = fmaf(sw2[o], h1, acc);
  }
  acc += b2[0];
  att[(size_t)b * (H_ * W_) + pix] = 1.0f / (1.0f + __expf(-acc));
}

// --------------------- K5: out = x * att (streaming) ------------------------
__global__ __launch_bounds__(256) void k5_modulate(
    const f4* __restrict__ x4, const f4* __restrict__ att4,
    f4* __restrict__ out4)
{
  int tid = blockIdx.x * 256 + threadIdx.x;   // 16*256*4096 = 16,777,216 float4
  int s4 = tid & 4095;                        // spatial / 4
  int bc = tid >> 12;
  int b  = bc >> 8;
  f4 xv = __builtin_nontemporal_load(&x4[tid]);        // x: read once, bypass
  f4 av = att4[b * 4096 + s4];                          // att: 1 MB, L2-hot
  f4 r  = xv * av;
  __builtin_nontemporal_store(r, &out4[tid]);          // out: write once
}

// ---------------------------------------------------------------------------
extern "C" void kernel_launch(void* const* d_in, const int* in_sizes, int n_in,
                              void* d_out, int out_size, void* d_ws, size_t ws_size,
                              hipStream_t stream) {
  const float* x  = (const float*)d_in[0];   // (16,256,128,128)
  const float* wp = (const float*)d_in[1];   // (256,64,3,3)
  const float* bp = (const float*)d_in[2];   // (256,)
  const float* w1 = (const float*)d_in[3];   // (64,256)
  const float* b1 = (const float*)d_in[4];   // (64,)
  const float* w2 = (const float*)d_in[5];   // (1,64)
  const float* b2 = (const float*)d_in[6];   // (1,)
  float* out = (float*)d_out;

  char* ws = (char*)d_ws;
  float* pf  = (float*)(ws);                               // 2 MB
  float* pp  = (float*)(ws + (size_t)2 * 1024 * 1024);     // 2 MB
  float* q   = (float*)(ws + (size_t)4 * 1024 * 1024);     // 512 KB
  float* att = (float*)(ws + (size_t)4 * 1024 * 1024 + 512 * 1024); // 1 MB

  // K1: x -> pf          (524288 threads)
  k1_cart2polar<<<(B_ * C_ * P_) / 256, 256, 0, stream>>>(x, pf);
  // K2: pf -> pp         (one block per (b, group))
  k2_polar_conv<<<B_ * 4, 256, 0, stream>>>(pf, wp, bp, pp);
  // K3: pp -> q          (one wave per 16x16 tile: 16 * 4 * 8 = 512 waves)
  k3_attn1_wmma<<<B_ * 4 * 8, 32, 0, stream>>>(pp, w1, q);
  // K4: q -> att         (16 batches * 64 blocks)
  k4_att<<<B_ * 64, 256, 0, stream>>>(q, b1, w2, b2, att);
  // K5: out = x * att    (16.7M float4)
  k5_modulate<<<(B_ * C_ * H_ * W_ / 4) / 256, 256, 0, stream>>>(
      (const f4*)x, (const f4*)att, (f4*)out);
}
